// MultiHeadAttention_18038862643467
// MI455X (gfx1250) — compile-verified, matrix operand order checked by reading, not run on hardware
//
#include <hip/hip_runtime.h>
#include <hip/hip_bf16.h>
#include <cstdint>

typedef unsigned short u16;
typedef __attribute__((ext_vector_type(16))) __bf16        v16bf;
typedef __attribute__((ext_vector_type(8)))  float         v8f;
typedef __attribute__((ext_vector_type(4))) unsigned int   u32x4;
typedef __attribute__((ext_vector_type(8)))  int           i32x8;
typedef __attribute__((ext_vector_type(4)))  int           i32x4;

#define EMBED 1024
#define SEQ   2048
#define BATCH 4
#define HEADS 16
#define HDIM  64
#define MROWS (BATCH*SEQ)      /* 8192 */
#define PS_STRIDE 2056         /* 2048 + 8 halves padding -> bank spread */

#define PANEL_STRIDE 40                     /* halves per LDS panel row: 32 + 8 pad */
#define PANEL_ROWS   128
#define PANEL_HALVES (PANEL_ROWS*PANEL_STRIDE)   /* 5120 halves = 10240 B */

// ---------- helpers ----------

__device__ __forceinline__ u16 f2bf(float f) {   // round-to-nearest-even f32->bf16
    union { float f; unsigned int u; } a; a.f = f;
    unsigned int u = a.u;
    unsigned int r = (u + 0x7FFFu + ((u >> 16) & 1u)) >> 16;
    return (u16)r;
}

__device__ __forceinline__ v8f zero8() {
    v8f z;
#pragma unroll
    for (int i = 0; i < 8; ++i) z[i] = 0.0f;
    return z;
}

// Load one 16x32 bf16 operand tile (A layout; B mirrors with N<->M).
// Lane holds row (lane&15); K chunks [8h,8h+8) and [16+8h,16+8h+8), h=lane>>4.
__device__ __forceinline__ v16bf load_tile(const u16* base, int stride, int lane) {
    const u16* p = base + (size_t)(lane & 15) * stride + 8 * (lane >> 4);
    union { v16bf v; u32x4 u[2]; } t;
    t.u[0] = *(const u32x4*)(p);
    t.u[1] = *(const u32x4*)(p + 16);
    return t.v;
}

__device__ __forceinline__ v8f wmma_bf16(v16bf a, v16bf b, v8f c) {
    return __builtin_amdgcn_wmma_f32_16x16x32_bf16(false, a, false, b, (short)0, c, false, false);
}

// Low 32 bits of a generic pointer to LDS == LDS byte offset (flat->LDS truncation).
__device__ __forceinline__ unsigned lds_off(const void* p) {
    return (unsigned)(uintptr_t)p;
}

// TDM: async-load a 2D tile (tile_dim1=128 rows x tile_dim0=32 bf16) from a
// row-major tensor (row length = stride = tensor_dim0 elements) into LDS,
// padding each 64B row with 16B so the LDS row stride is PANEL_STRIDE halves.
__device__ __forceinline__ void tdm_load_panel(unsigned lds_byte_addr,
                                               const u16* gaddr,
                                               unsigned row_len /* elements */) {
    const unsigned long long ga = (unsigned long long)(uintptr_t)gaddr;
    u32x4 g0;
    g0[0] = 1u;                                            // count=1 (valid user D#)
    g0[1] = lds_byte_addr;                                 // lds_addr [63:32]
    g0[2] = (unsigned)(ga & 0xFFFFFFFFu);                  // global_addr [95:64]
    g0[3] = (unsigned)((ga >> 32) & 0x01FFFFFFu)           // global_addr [120:96]
            | (2u << 30);                                  // type=2 ("image")
    i32x8 g1;
    g1[0] = (int)((1u << 16)        /* data_size = 2 bytes            */
                | (1u << 20)        /* pad_enable                     */
                | (3u << 22)        /* pad_interval: 16 DWORDs (64B)  */
                | (3u << 25));      /* pad_amount:   4 DWORDs (16B)   */
    g1[1] = (int)((row_len & 0xFFFFu) << 16);              // tensor_dim0 [15:0]
    g1[2] = (int)((row_len >> 16) | ((unsigned)MROWS << 16)); // dim0 hi | tensor_dim1 lo
    g1[3] = (int)(32u << 16);                              // tensor_dim1 hi=0 | tile_dim0=32
    g1[4] = (int)128u;                                     // tile_dim1=128, tile_dim2=0
    g1[5] = (int)row_len;                                  // tensor_dim0_stride [31:0]
    g1[6] = 0;                                             // stride hi | dim1_stride lo
    g1[7] = 0;                                             // dim1_stride hi
    const i32x4 gz4 = {0, 0, 0, 0};                        // groups 2/3: dims unused (2D)
    const i32x8 gz8 = {0, 0, 0, 0, 0, 0, 0, 0};            // clang-23 extra operand
    __builtin_amdgcn_tensor_load_to_lds(g0, g1, gz4, gz4, gz8, 0);
}

// ---------- kernels ----------

__global__ void cvt_f32_to_bf16(const float* __restrict__ src, u16* __restrict__ dst, int n) {
    int i = blockIdx.x * blockDim.x + threadIdx.x;
    if (i < n) dst[i] = f2bf(src[i]);
}

// C[M,N] = Xbf16[M,K] @ Wbf16[N,K]^T + bias[N], TDM-staged panels in LDS.
// mode 0: store bf16 head-split  [B,H,S,D]
// mode 1: store bf16 head-split transposed [B,H,D,S]
// mode 2: store f32 row-major [M,N]
// grid = (N/128, M/128), block = 256 (8 waves, 2x4 wave grid, 64x32 per wave)
__global__ __launch_bounds__(256) void gemm_bf16_nt(
    const u16* __restrict__ X, const u16* __restrict__ W,
    const float* __restrict__ bias, void* __restrict__ outp,
    int K, int N, int mode)
{
    __shared__ __align__(16) u16 sx[2 * PANEL_HALVES];   // X panels (double buffered)
    __shared__ __align__(16) u16 sw[2 * PANEL_HALVES];   // W panels

    const int lane = threadIdx.x & 31;
    const int wave = threadIdx.x >> 5;
    const int wm = wave >> 2, wn = wave & 3;
    const int rowWG = blockIdx.y * 128;
    const int colWG = blockIdx.x * 128;
    const int r = lane & 15, h = lane >> 4;

    v8f acc[4][2];
#pragma unroll
    for (int mi = 0; mi < 4; ++mi)
#pragma unroll
        for (int ni = 0; ni < 2; ++ni) acc[mi][ni] = zero8();

    const int KT = K >> 5;
    if (wave == 0) {
        tdm_load_panel(lds_off(sx), X + (size_t)rowWG * K, (unsigned)K);
        tdm_load_panel(lds_off(sw), W + (size_t)colWG * K, (unsigned)K);
    }

    for (int kt = 0; kt < KT; ++kt) {
        const int buf = kt & 1;
        if (wave == 0) {
            if (kt + 1 < KT) {      // prefetch next panel pair, then wait for current
                const int nb = buf ^ 1;
                const int kn = (kt + 1) << 5;
                tdm_load_panel(lds_off(sx + nb * PANEL_HALVES),
                               X + (size_t)rowWG * K + kn, (unsigned)K);
                tdm_load_panel(lds_off(sw + nb * PANEL_HALVES),
                               W + (size_t)colWG * K + kn, (unsigned)K);
                __builtin_amdgcn_s_wait_tensorcnt(2);
            } else {
                __builtin_amdgcn_s_wait_tensorcnt(0);
            }
        }
        __syncthreads();            // panels for `buf` are visible to all waves

        const u16* ax = sx + buf * PANEL_HALVES + (wm * 64) * PANEL_STRIDE;
        const u16* bw = sw + buf * PANEL_HALVES + (wn * 32) * PANEL_STRIDE;
        v16bf a[4], b[2];
#pragma unroll
        for (int mi = 0; mi < 4; ++mi)
            a[mi] = load_tile(ax + mi * 16 * PANEL_STRIDE, PANEL_STRIDE, lane);
#pragma unroll
        for (int ni = 0; ni < 2; ++ni)
            b[ni] = load_tile(bw + ni * 16 * PANEL_STRIDE, PANEL_STRIDE, lane);
#pragma unroll
        for (int mi = 0; mi < 4; ++mi)
#pragma unroll
            for (int ni = 0; ni < 2; ++ni)
                acc[mi][ni] = wmma_bf16(a[mi], b[ni], acc[mi][ni]);

        __syncthreads();            // everyone done reading before `buf` is overwritten
    }

#pragma unroll
    for (int mi = 0; mi < 4; ++mi) {
#pragma unroll
        for (int ni = 0; ni < 2; ++ni) {
            const int col = colWG + (wn * 32) + ni * 16 + r;
            const float bv = bias[col];
#pragma unroll
            for (int i = 0; i < 8; ++i) {
                const int row = rowWG + (wm * 64) + mi * 16 + i + 8 * h;
                const float v = acc[mi][ni][i] + bv;
                if (mode == 2) {
                    ((float*)outp)[(size_t)row * N + col] = v;
                } else {
                    const int bb = row >> 11, s = row & 2047;   // S = 2048
                    const int hh = col >> 6,  d = col & 63;     // D = 64
                    u16* o = (u16*)outp;
                    if (mode == 0)
                        o[(((size_t)(bb * HEADS + hh)) * SEQ + s) * HDIM + d] = f2bf(v);
                    else
                        o[(((size_t)(bb * HEADS + hh)) * HDIM + d) * SEQ + s] = f2bf(v);
                }
            }
        }
    }
}

// One workgroup per (b*H+h, 16-row q block). 8 waves x 256 columns each.
// Emits normalized weights (f32) to wout and bf16 merged ctx to ctxb.
__global__ __launch_bounds__(256) void attn_fused(
    const u16* __restrict__ Qh, const u16* __restrict__ Kh, const u16* __restrict__ Vt,
    float* __restrict__ wout, u16* __restrict__ ctxb)
{
    __shared__ __align__(16) u16 Ps[16 * PS_STRIDE];   // softmax probs, bf16
    __shared__ float red[16][8];                        // cross-wave row reduce
    __shared__ float ctxs[16][64];                      // ctx accumulation

    const int tid  = threadIdx.x;
    const int lane = tid & 31;
    const int wave = tid >> 5;
    const int r = lane & 15, h = lane >> 4;
    const int bh    = blockIdx.x >> 7;           // b*H + h
    const int qbase = (blockIdx.x & 127) * 16;
    const int wcol  = wave * 256;

    const u16* Qrows = Qh + ((size_t)bh * SEQ + qbase) * HDIM;
    const u16* Krows = Kh + (size_t)bh * SEQ * HDIM;
    const u16* Vrows = Vt + (size_t)bh * HDIM * SEQ;

    const v16bf qa0 = load_tile(Qrows,      HDIM, lane);
    const v16bf qa1 = load_tile(Qrows + 32, HDIM, lane);

    // ---- scores: 16 rows x 256 cols per wave, kept in registers ----
    v8f sacc[16];
#pragma unroll
    for (int ct = 0; ct < 16; ++ct) sacc[ct] = zero8();

#pragma unroll 4
    for (int ct = 0; ct < 16; ++ct) {
        const u16* Kt = Krows + (size_t)(wcol + ct * 16) * HDIM;
        const v16bf kb0 = load_tile(Kt,      HDIM, lane);
        const v16bf kb1 = load_tile(Kt + 32, HDIM, lane);
        sacc[ct] = wmma_bf16(qa0, kb0, sacc[ct]);
        sacc[ct] = wmma_bf16(qa1, kb1, sacc[ct]);
    }

    // ---- softmax: scale, rowmax, exp, rowsum, normalize ----
    const float scale = 0.125f;                // 1/sqrt(64)
    float m8[8];
#pragma unroll
    for (int i = 0; i < 8; ++i) m8[i] = -3.0e38f;
#pragma unroll
    for (int ct = 0; ct < 16; ++ct)
#pragma unroll
        for (int i = 0; i < 8; ++i) {
            const float v = sacc[ct][i] * scale;
            sacc[ct][i] = v;
            m8[i] = fmaxf(m8[i], v);
        }
#pragma unroll
    for (int i = 0; i < 8; ++i)
        for (int off = 1; off < 16; off <<= 1)
            m8[i] = fmaxf(m8[i], __shfl_xor(m8[i], off, 16));
    if (r == 0) {
#pragma unroll
        for (int i = 0; i < 8; ++i) red[i + 8 * h][wave] = m8[i];
    }
    __syncthreads();
    float gM[8];
#pragma unroll
    for (int i = 0; i < 8; ++i) {
        float mm = -3.0e38f;
#pragma unroll
        for (int w2 = 0; w2 < 8; ++w2) mm = fmaxf(mm, red[i + 8 * h][w2]);
        gM[i] = mm;
    }
    __syncthreads();

    float s8[8];
#pragma unroll
    for (int i = 0; i < 8; ++i) s8[i] = 0.0f;
#pragma unroll
    for (int ct = 0; ct < 16; ++ct)
#pragma unroll
        for (int i = 0; i < 8; ++i) {
            const float p = __expf(sacc[ct][i] - gM[i]);
            sacc[ct][i] = p;
            s8[i] += p;
        }
#pragma unroll
    for (int i = 0; i < 8; ++i)
        for (int off = 1; off < 16; off <<= 1)
            s8[i] += __shfl_xor(s8[i], off, 16);
    if (r == 0) {
#pragma unroll
        for (int i = 0; i < 8; ++i) red[i + 8 * h][wave] = s8[i];
    }
    __syncthreads();
    float gInv[8];
#pragma unroll
    for (int i = 0; i < 8; ++i) {
        float ss = 0.0f;
#pragma unroll
        for (int w2 = 0; w2 < 8; ++w2) ss += red[i + 8 * h][w2];
        gInv[i] = 1.0f / ss;
    }

    // ---- write weights (f32 output) + bf16 probs to LDS ----
#pragma unroll 4
    for (int ct = 0; ct < 16; ++ct) {
        const int col = wcol + ct * 16 + r;
#pragma unroll
        for (int i = 0; i < 8; ++i) {
            const int row = i + 8 * h;
            const float p = sacc[ct][i] * gInv[i];
            wout[((size_t)bh * SEQ + qbase + row) * SEQ + col] = p;
            Ps[row * PS_STRIDE + col] = f2bf(p);
        }
    }
    for (int idx = tid; idx < 16 * 64; idx += 256) ((float*)ctxs)[idx] = 0.0f;
    __syncthreads();

    // ---- ctx = P @ V : each wave covers its 256-wide K slice, reduce via LDS atomics ----
    for (int n = 0; n < 4; ++n) {
        v8f cacc = zero8();
#pragma unroll
        for (int kt = 0; kt < 8; ++kt) {
            const int k0 = wcol + kt * 32;
            const v16bf pa = load_tile(Ps + k0, PS_STRIDE, lane);
            const v16bf vb = load_tile(Vrows + (size_t)(n * 16) * SEQ + k0, SEQ, lane);
            cacc = wmma_bf16(pa, vb, cacc);
        }
        const int col = n * 16 + r;
#pragma unroll
        for (int i = 0; i < 8; ++i) atomicAdd(&ctxs[i + 8 * h][col], cacc[i]);
    }
    __syncthreads();

    // ---- write merged ctx (bf16) [B,S,E] ----
    const int b = bh >> 4, hh = bh & 15;
    for (int idx = tid; idx < 1024; idx += 256) {
        const int row = idx >> 6, d = idx & 63;
        ctxb[((size_t)b * SEQ + qbase + row) * EMBED + hh * HDIM + d] = f2bf(ctxs[row][d]);
    }
}

// ---------- host launcher ----------

extern "C" void kernel_launch(void* const* d_in, const int* in_sizes, int n_in,
                              void* d_out, int out_size, void* d_ws, size_t ws_size,
                              hipStream_t stream) {
    (void)in_sizes; (void)n_in; (void)out_size; (void)ws_size;
    const float* q  = (const float*)d_in[0];
    const float* k  = (const float*)d_in[1];
    const float* v  = (const float*)d_in[2];
    const float* Wq = (const float*)d_in[3];
    const float* bq = (const float*)d_in[4];
    const float* Wk = (const float*)d_in[5];
    const float* bk = (const float*)d_in[6];
    const float* Wv = (const float*)d_in[7];
    const float* bv = (const float*)d_in[8];
    const float* Wo = (const float*)d_in[9];
    const float* bo = (const float*)d_in[10];

    float* out_f = (float*)d_out;                         // [B,S,E]
    float* wts   = out_f + (size_t)MROWS * EMBED;         // [B,H,S,S]

    char* ws = (char*)d_ws;
    size_t off = 0;
    auto take = [&](size_t bytes) -> void* {
        void* p = ws + off;
        off += (bytes + 255) & ~(size_t)255;
        return p;
    };
    const size_t nBig = (size_t)MROWS * EMBED;            // 8,388,608
    const size_t nW   = (size_t)EMBED * EMBED;            // 1,048,576

    u16* qb   = (u16*)take(nBig * 2);
    u16* kb   = (u16*)take(nBig * 2);
    u16* vb   = (u16*)take(nBig * 2);
    u16* Wqb  = (u16*)take(nW * 2);
    u16* Wkb  = (u16*)take(nW * 2);
    u16* Wvb  = (u16*)take(nW * 2);
    u16* Wob  = (u16*)take(nW * 2);
    u16* Qhd  = (u16*)take(nBig * 2);   // [B,H,S,D]
    u16* Khd  = (u16*)take(nBig * 2);   // [B,H,S,D]
    u16* Vtd  = (u16*)take(nBig * 2);   // [B,H,D,S]
    u16* ctxb = (u16*)take(nBig * 2);   // [B,S,E]

    const dim3 blk(256);
    const int gBig = (int)((nBig + 255) / 256);
    const int gW   = (int)((nW + 255) / 256);

    cvt_f32_to_bf16<<<gBig, blk, 0, stream>>>(q,  qb,  (int)nBig);
    cvt_f32_to_bf16<<<gBig, blk, 0, stream>>>(k,  kb,  (int)nBig);
    cvt_f32_to_bf16<<<gBig, blk, 0, stream>>>(v,  vb,  (int)nBig);
    cvt_f32_to_bf16<<<gW,   blk, 0, stream>>>(Wq, Wqb, (int)nW);
    cvt_f32_to_bf16<<<gW,   blk, 0, stream>>>(Wk, Wkb, (int)nW);
    cvt_f32_to_bf16<<<gW,   blk, 0, stream>>>(Wv, Wvb, (int)nW);
    cvt_f32_to_bf16<<<gW,   blk, 0, stream>>>(Wo, Wob, (int)nW);

    const dim3 ggrid(EMBED / 128, MROWS / 128);           // (8, 64)
    gemm_bf16_nt<<<ggrid, blk, 0, stream>>>(qb, Wqb, bq, Qhd, EMBED, EMBED, 0);
    gemm_bf16_nt<<<ggrid, blk, 0, stream>>>(kb, Wkb, bk, Khd, EMBED, EMBED, 0);
    gemm_bf16_nt<<<ggrid, blk, 0, stream>>>(vb, Wvb, bv, Vtd, EMBED, EMBED, 1);

    attn_fused<<<dim3(BATCH * HEADS * (SEQ / 16)), blk, 0, stream>>>(Qhd, Khd, Vtd, wts, ctxb);

    gemm_bf16_nt<<<ggrid, blk, 0, stream>>>(ctxb, Wob, bo, out_f, EMBED, EMBED, 2);
}
